// DualConsensusNet_18588618457439
// MI455X (gfx1250) — matrix operand order repaired and checked by reading, not compiled, not run on hardware
//
#include <hip/hip_runtime.h>
#include <hip/hip_bf16.h>
#include <math.h>

// ---- problem constants (match reference) ----
#define NS_   4000
#define NT_   4096
#define E_S_  64000
#define E_T_  65536
#define DIN_  128
#define DH_   256
#define CAT_  896      // 128 + 3*256
#define SINK_N 4096
#define EPS_  1e-10f
#define ALPHA_ 20.0f

typedef __attribute__((ext_vector_type(16))) __bf16 v16bf;
typedef __attribute__((ext_vector_type(8)))  __bf16 v8bf;
typedef __attribute__((ext_vector_type(8)))  float  v8f;

#define LDP 40   // LDS row pitch in bf16 (80 bytes: 16B aligned, bank-spread)

// ----------------------------------------------------------------------------
// WMMA GEMM, all-bf16 operands, k-major both sides:
//   C[M,N] = A[M,K] @ Bt[N,K]^T  (+ bias if HB), C in f32.
// A: bf16 row-major (lda), Bt: bf16 row-major (ldb) holding B transposed.
// M % 32 == 0, N % 64 == 0, K % 32 == 0.
// Block = 128 threads = 4 waves computing a 32x64 C tile; each wave owns a
// 16x32 strip: one shared A fragment + two B fragments -> two
// v_wmma_f32_16x16x32_bf16 per K-step. Tiles and fragments are pure 16-byte
// moves: global_load_b128 -> ds_store_b128 staging, ds_load_b128 fragments.
// ----------------------------------------------------------------------------
template <int HB>
__global__ __launch_bounds__(128)
void gemm_bf16kt(const __bf16* __restrict__ A, const __bf16* __restrict__ Bt,
                 float* __restrict__ C, const float* __restrict__ bias,
                 int M, int N, int K, int lda, int ldb, int ldc)
{
    __shared__ __bf16 As[32][LDP];   // [m][k]
    __shared__ __bf16 Bs[64][LDP];   // [n][k]

    const int tid   = threadIdx.x;
    const int tileN = blockIdx.x * 64;
    const int tileM = blockIdx.y * 32;
    const int w     = tid >> 5;          // wave id 0..3
    const int lane  = tid & 31;
    const int wm    = (w >> 1) << 4;     // row group: 0 or 16
    const int wn    = (w & 1) << 5;      // col group: 0 or 32
    const int lm    = lane & 15;
    const int hh    = lane >> 4;         // lane half

    v8f acc0 = {};
    v8f acc1 = {};

    for (int k0 = 0; k0 < K; k0 += 32) {
        // ---- A tile: 32 rows x 32 bf16 = 128 x 16B chunks, 1 per thread ----
        {
            int r = tid >> 2, c8 = (tid & 3) << 3;
            *(v8bf*)&As[r][c8] =
                *(const v8bf*)&A[(size_t)(tileM + r) * lda + (k0 + c8)];
        }
        // ---- B tile: 64 rows x 32 bf16 = 256 x 16B chunks, 2 per thread ----
        #pragma unroll
        for (int i = 0; i < 2; ++i) {
            int l = tid + (i << 7);
            int r = l >> 2, c8 = (l & 3) << 3;
            *(v8bf*)&Bs[r][c8] =
                *(const v8bf*)&Bt[(size_t)(tileN + r) * ldb + (k0 + c8)];
        }
        __syncthreads();

        // ---- fragments: each = two 16B ds_load_b128 (contiguous k = 16*hh + s, same order for A and B) ----
        union { v16bf v; v8bf h[2]; } af, bf0, bf1;
        const __bf16* arow  = &As[wm + lm][0];
        af.h[0] = *(const v8bf*)(arow + (hh << 4));
        af.h[1] = *(const v8bf*)(arow + (hh << 4) + 8);
        const __bf16* brow0 = &Bs[wn + lm][hh << 4];
        bf0.h[0] = *(const v8bf*)(brow0);
        bf0.h[1] = *(const v8bf*)(brow0 + 8);
        const __bf16* brow1 = &Bs[wn + 16 + lm][hh << 4];
        bf1.h[0] = *(const v8bf*)(brow1);
        bf1.h[1] = *(const v8bf*)(brow1 + 8);

        acc0 = __builtin_amdgcn_wmma_f32_16x16x32_bf16(
            false, af.v, false, bf0.v, (short)0, acc0, false, false);
        acc1 = __builtin_amdgcn_wmma_f32_16x16x32_bf16(
            false, af.v, false, bf1.v, (short)0, acc1, false, false);
        __syncthreads();
    }

    // C/D layout: VGPR v holds row m = hh*8 + v, col n = lm (within 16x16)
    const int col0 = tileN + wn + lm;
    #pragma unroll
    for (int v = 0; v < 8; ++v) {
        int row = tileM + wm + (hh << 3) + v;
        float o0 = acc0[v];
        float o1 = acc1[v];
        if (HB) { o0 += bias[col0]; o1 += bias[col0 + 16]; }
        C[(size_t)row * ldc + col0]      = o0;
        C[(size_t)row * ldc + col0 + 16] = o1;
    }
}

// ---- helpers ---------------------------------------------------------------
__global__ void fill_zero_k(float* __restrict__ p, size_t n) {
    size_t i = (size_t)blockIdx.x * blockDim.x + threadIdx.x;
    if (i < n) p[i] = 0.0f;
}

// transpose f32 weight [K,N] -> bf16 [N,K] (tiny, one-shot)
__global__ void wtrans_bf16_k(const float* __restrict__ W, __bf16* __restrict__ Wt,
                              int K, int N) {
    int k = blockIdx.x * 256 + threadIdx.x;
    int n = blockIdx.y;
    if (k < K) Wt[(size_t)n * K + k] = (__bf16)W[(size_t)k * N + n];
}

// copy x (N x 128) into the first 128 columns of bf16 concat buffer (ld=896)
__global__ void copy_x_k(const float* __restrict__ x, __bf16* __restrict__ cat) {
    int n = blockIdx.x, c = threadIdx.x;   // 128 threads
    cat[(size_t)n * CAT_ + c] = (__bf16)x[(size_t)n * DIN_ + c];
}

// edge-parallel scatter add: accum[to[e]][:] += P[from[e]][:], cnt[to[e]] += 1
__global__ void scatter_add_k(const float* __restrict__ P,
                              const int* __restrict__ from,
                              const int* __restrict__ to,
                              float* __restrict__ accum,
                              float* __restrict__ cnt) {
    int e = blockIdx.x;
    int c = threadIdx.x;                   // 256 threads
    int f = from[e], t = to[e];
    atomicAdd(&accum[(size_t)t * DH_ + c], P[(size_t)f * DH_ + c]);
    if (c == 0) atomicAdd(&cnt[t], 1.0f);
}

// h = relu(Pr + br + agg1/max(cnt1,1) + agg2/max(cnt2,1)) -> bf16 concat seg
__global__ void combine_relu_k(const float* __restrict__ Pr,
                               const float* __restrict__ br,
                               const float* __restrict__ agg1,
                               const float* __restrict__ cnt1,
                               const float* __restrict__ agg2,
                               const float* __restrict__ cnt2,
                               __bf16* __restrict__ outSeg) {
    int n = blockIdx.x, c = threadIdx.x;   // 256 threads
    size_t g = (size_t)n * DH_ + c;
    float c1 = fmaxf(cnt1[n], 1.0f);
    float c2 = fmaxf(cnt2[n], 1.0f);
    float v = Pr[g] + br[c] + agg1[g] / c1 + agg2[g] / c2;
    outSeg[(size_t)n * CAT_ + c] = (__bf16)fmaxf(v, 0.0f);
}

// row l2-normalize f32 input, emit bf16 embedding for the match GEMM
__global__ void l2norm_bf16_k(const float* __restrict__ h, __bf16* __restrict__ hb) {
    __shared__ float red[256];
    int n = blockIdx.x, c = threadIdx.x;   // 256 threads, one row per block
    float v = h[(size_t)n * DH_ + c];
    red[c] = v * v;
    __syncthreads();
    for (int s = 128; s > 0; s >>= 1) {
        if (c < s) red[c] += red[c + s];
        __syncthreads();
    }
    float norm = fmaxf(sqrtf(red[0]), 1e-12f);
    hb[(size_t)n * DH_ + c] = (__bf16)(v / norm);
}

// S[r][c] = exp(alpha * ((r<NS ? S[r][c] : EPS) + EPS))   over 4096x4096
__global__ void sink_init_k(float* __restrict__ S) {
    size_t g = (size_t)blockIdx.x * 256 + threadIdx.x;
    int r = (int)(g >> 12);                 // /4096
    float base = (r < NS_) ? S[g] : EPS_;
    S[g] = expf(ALPHA_ * (base + EPS_));
}

__global__ void col_sum_k(const float* __restrict__ S, float* __restrict__ sums) {
    int c = blockIdx.x * 256 + threadIdx.x;  // 4096 threads total
    float s = 0.0f;
    for (int r = 0; r < SINK_N; ++r) {
        if (r + 16 < SINK_N)
            __builtin_prefetch(&S[(size_t)(r + 16) * SINK_N + c], 0, 1);
        s += S[(size_t)r * SINK_N + c];
    }
    sums[c] = s;
}

__global__ void row_sum_k(const float* __restrict__ S, float* __restrict__ sums) {
    __shared__ float red[256];
    int r = blockIdx.x, c = threadIdx.x;
    float s = 0.0f;
    for (int i = c; i < SINK_N; i += 256) s += S[(size_t)r * SINK_N + i];
    red[c] = s;
    __syncthreads();
    for (int st = 128; st > 0; st >>= 1) {
        if (c < st) red[c] += red[c + st];
        __syncthreads();
    }
    if (c == 0) sums[r] = red[0];
}

__global__ void div_col_k(float* __restrict__ S, const float* __restrict__ sums) {
    size_t g = (size_t)blockIdx.x * 256 + threadIdx.x;
    S[g] /= sums[g & (SINK_N - 1)];
}

__global__ void div_row_k(float* __restrict__ S, const float* __restrict__ sums) {
    size_t g = (size_t)blockIdx.x * 256 + threadIdx.x;
    S[g] /= sums[g >> 12];
}

// ----------------------------------------------------------------------------
extern "C" void kernel_launch(void* const* d_in, const int* in_sizes, int n_in,
                              void* d_out, int out_size, void* d_ws, size_t ws_size,
                              hipStream_t stream) {
    (void)in_sizes; (void)n_in; (void)out_size; (void)ws_size;

    const float* x_s = (const float*)d_in[0];
    const float* x_t = (const float*)d_in[1];
    const int*   e_s = (const int*)d_in[2];
    const int*   e_t = (const int*)d_in[3];
    const float* final_b = (const float*)d_in[17];

    // ---- workspace layout: f32 region first, then 16B-aligned bf16 region ----
    float* ws = (float*)d_ws;
    size_t o = 0;
    float* hf_s  = ws + o; o += (size_t)NS_ * DH_;
    float* hf_t  = ws + o; o += (size_t)NT_ * DH_;
    float* P     = ws + o; o += (size_t)NT_ * DH_ * 3;   // P1 | P2 | Pr
    float* agg1  = ws + o; o += (size_t)NT_ * DH_;
    float* agg2  = ws + o; o += (size_t)NT_ * DH_;
    float* cnt1  = ws + o; o += NT_;
    float* cnt2  = ws + o; o += NT_;
    float* Spad  = ws + o; o += (size_t)SINK_N * SINK_N;
    float* sums  = ws + o; o += SINK_N;

    __bf16* bb = (__bf16*)(ws + o);          // o*4 bytes is 16B aligned
    size_t ob = 0;
    __bf16* cat_s = bb + ob; ob += (size_t)NS_ * CAT_;
    __bf16* cat_t = bb + ob; ob += (size_t)NT_ * CAT_;
    __bf16* hb_s  = bb + ob; ob += (size_t)NS_ * DH_;
    __bf16* hb_t  = bb + ob; ob += (size_t)NT_ * DH_;
    __bf16* Wt[9];                           // transposed bf16 layer weights
    for (int l = 0; l < 3; ++l) {
        int fan = (l == 0) ? DIN_ : DH_;
        for (int j = 0; j < 3; ++j) { Wt[l * 3 + j] = bb + ob; ob += (size_t)DH_ * fan; }
    }
    __bf16* fwt = bb + ob; ob += (size_t)DH_ * CAT_;     // [256, 896]

    // ---- one-shot: transpose+convert all weights to bf16 [N,K] ----
    for (int l = 0; l < 3; ++l) {
        int fan = (l == 0) ? DIN_ : DH_;
        for (int j = 0; j < 3; ++j) {
            const float* W = (const float*)d_in[4 + l * 4 + j];   // [fan, 256]
            wtrans_bf16_k<<<dim3((fan + 255) / 256, DH_), 256, 0, stream>>>(
                W, Wt[l * 3 + j], fan, DH_);
        }
    }
    wtrans_bf16_k<<<dim3((CAT_ + 255) / 256, DH_), 256, 0, stream>>>(
        (const float*)d_in[16], fwt, CAT_, DH_);

    // ---- one RelCNN pass ----
    auto run_graph = [&](const float* x, const int* edge, int N, int E,
                         __bf16* cat, float* hf, __bf16* hb) {
        copy_x_k<<<N, DIN_, 0, stream>>>(x, cat);
        int fan = DIN_, off = 0;
        for (int l = 0; l < 3; ++l) {
            const float* br = (const float*)d_in[4 + l * 4 + 3];
            const __bf16* h = cat + off;              // N x fan, ld = CAT_
            float* P1 = P;
            float* P2 = P + (size_t)N * DH_;
            float* Pr = P + 2 * (size_t)N * DH_;
            dim3 g(DH_ / 64, N / 32);
            gemm_bf16kt<0><<<g, 128, 0, stream>>>(h, Wt[l * 3 + 0], P1, nullptr,
                N, DH_, fan, CAT_, fan, DH_);
            gemm_bf16kt<0><<<g, 128, 0, stream>>>(h, Wt[l * 3 + 1], P2, nullptr,
                N, DH_, fan, CAT_, fan, DH_);
            gemm_bf16kt<0><<<g, 128, 0, stream>>>(h, Wt[l * 3 + 2], Pr, nullptr,
                N, DH_, fan, CAT_, fan, DH_);

            size_t na = (size_t)N * DH_;
            fill_zero_k<<<(unsigned)((na + 255) / 256), 256, 0, stream>>>(agg1, na);
            fill_zero_k<<<(unsigned)((na + 255) / 256), 256, 0, stream>>>(agg2, na);
            fill_zero_k<<<(unsigned)((N + 255) / 256), 256, 0, stream>>>(cnt1, (size_t)N);
            fill_zero_k<<<(unsigned)((N + 255) / 256), 256, 0, stream>>>(cnt2, (size_t)N);

            const int* src = edge;          // edge[0,:]
            const int* dst = edge + E;      // edge[1,:]
            scatter_add_k<<<E, DH_, 0, stream>>>(P1, src, dst, agg1, cnt1);
            scatter_add_k<<<E, DH_, 0, stream>>>(P2, dst, src, agg2, cnt2);

            off += (l == 0) ? DIN_ : DH_;   // next segment start: 128, 384, 640
            combine_relu_k<<<N, DH_, 0, stream>>>(Pr, br, agg1, cnt1, agg2, cnt2,
                                                  cat + off);
            fan = DH_;
        }
        // final projection: hf = cat (N x 896) @ final_w + final_b
        gemm_bf16kt<1><<<dim3(DH_ / 64, N / 32), 128, 0, stream>>>(
            cat, fwt, hf, final_b, N, DH_, CAT_, CAT_, CAT_, DH_);
        l2norm_bf16_k<<<N, DH_, 0, stream>>>(hf, hb);
    };

    run_graph(x_s, e_s, NS_, E_S_, cat_s, hf_s, hb_s);
    run_graph(x_t, e_t, NT_, E_T_, cat_t, hf_t, hb_t);

    // ---- match = l2norm(h_s) @ l2norm(h_t)^T  -> rows [0,4000) of Spad ----
    gemm_bf16kt<0><<<dim3(SINK_N / 64, NS_ / 32), 128, 0, stream>>>(
        hb_s, hb_t, Spad, nullptr, NS_, SINK_N, DH_, DH_, DH_, SINK_N);

    // ---- Sinkhorn over padded 4096x4096 (fits in 192MB L2) ----
    const unsigned NELEM_BLKS = (SINK_N * (unsigned)SINK_N) / 256;  // 65536
    sink_init_k<<<NELEM_BLKS, 256, 0, stream>>>(Spad);
    for (int i = 0; i < 10; ++i) {
        if (i % 2 == 1) {   // row normalization
            row_sum_k<<<SINK_N, 256, 0, stream>>>(Spad, sums);
            div_row_k<<<NELEM_BLKS, 256, 0, stream>>>(Spad, sums);
        } else {            // column normalization
            col_sum_k<<<SINK_N / 256, 256, 0, stream>>>(Spad, sums);
            div_col_k<<<NELEM_BLKS, 256, 0, stream>>>(Spad, sums);
        }
    }

    // ---- drop dummy rows: first 4000 rows are contiguous ----
    hipMemcpyAsync(d_out, Spad, (size_t)NS_ * SINK_N * sizeof(float),
                   hipMemcpyDeviceToDevice, stream);
}
